// CLVPConditioningEncoder_68633577390142
// MI455X (gfx1250) — matrix operand order repaired and checked by reading, not compiled, not run on hardware
//
#include <hip/hip_runtime.h>

// ---------------------------------------------------------------------------
// Types for CDNA5 WMMA (wave32): V_WMMA_F32_16X16X32_BF16
// ---------------------------------------------------------------------------
typedef __attribute__((ext_vector_type(8)))  float  v8f;
typedef __attribute__((ext_vector_type(8)))  __bf16 v8bf;
typedef __attribute__((ext_vector_type(16))) __bf16 v16bf;

static __device__ inline v8f wmma_bf16(v16bf a, v16bf b, v8f c) {
  // D = A(16x32 bf16) * B(32x16 bf16) + C(16x16 f32)
  return __builtin_amdgcn_wmma_f32_16x16x32_bf16(
      /*neg_a=*/false, a, /*neg_b=*/false, b,
      /*c_mod=*/(short)0, c, /*reuse_a=*/false, /*reuse_b=*/false);
}

// A-matrix fragment (16x32, bf16), row-major source with leading dim `ld`.
// Contiguous K order: lane m=l&15, half=l>>4; elems 0..15 -> K = half*16 + 0..15.
static __device__ inline v16bf load_a_frag(const __bf16* base, int ld, int lane) {
  int m = lane & 15, half = lane >> 4;
  const __bf16* p = base + (size_t)m * ld + half * 16;
  v8bf lo = *(const v8bf*)p;
  v8bf hi = *(const v8bf*)(p + 8);
  return __builtin_shufflevector(lo, hi, 0,1,2,3,4,5,6,7,8,9,10,11,12,13,14,15);
}

// B-matrix fragment (32x16, bf16) from a K-contiguous (transposed) source
// Bt[n][k] with leading dim `ld`. Lane n=l&15, half=l>>4; elems = K = half*16+0..15.
static __device__ inline v16bf load_b_frag(const __bf16* base, int ld, int lane) {
  int n = lane & 15, half = lane >> 4;
  return *(const v16bf*)(base + (size_t)n * ld + half * 16);
}

// ---------------------------------------------------------------------------
// Problem constants
// ---------------------------------------------------------------------------
#define BATCH 2
#define SEQ   2048
#define EMB   1024
#define HEADS 16
#define HDIM  64
#define MROWS (BATCH * SEQ)       // 4096
#define NBUCK 32
#define TABW  (2 * SEQ - 1)       // 4095 relative positions

// ---------------------------------------------------------------------------
// Prep kernels
// ---------------------------------------------------------------------------
__global__ void convert_f32_to_bf16(const float* __restrict__ in,
                                    __bf16* __restrict__ out, int n) {
  int i = blockIdx.x * blockDim.x + threadIdx.x;
  if (i < n) out[i] = (__bf16)in[i];
}

// Wt[n][k] = (bf16) W[k][n], 1024x1024, tiled through LDS.
__global__ void transpose_w_bf16(const float* __restrict__ W,
                                 __bf16* __restrict__ Wt) {
  __shared__ float tile[32][33];
  int n0 = blockIdx.x * 32, k0 = blockIdx.y * 32;
  for (int dy = threadIdx.y; dy < 32; dy += 8)
    tile[dy][threadIdx.x] = W[(size_t)(k0 + dy) * EMB + n0 + threadIdx.x];
  __syncthreads();
  for (int dy = threadIdx.y; dy < 32; dy += 8)
    Wt[(size_t)(n0 + dy) * EMB + k0 + threadIdx.x] = (__bf16)tile[threadIdx.x][dy];
}

// tab[h][rel + 2047] = 8 * rel_bias[bucket(rel)][h]
__global__ void build_bias_table(const float* __restrict__ rel_bias,
                                 float* __restrict__ tab) {
  int idx = blockIdx.x * blockDim.x + threadIdx.x;
  if (idx >= HEADS * TABW) return;
  int h = idx / TABW, p = idx % TABW;
  int rel = p - (SEQ - 1);            // mem - ctx
  int rp = rel < 0 ? -rel : 0;        // max(-rel, 0)
  int bucket;
  if (rp < NBUCK / 2) {
    bucket = rp;
  } else {
    float l = logf((float)rp / 16.0f) * (1.0f / logf(4.0f)) * 16.0f;
    int lg = 16 + (int)l;
    bucket = lg < (NBUCK - 1) ? lg : (NBUCK - 1);
  }
  tab[(size_t)h * TABW + p] = 8.0f * rel_bias[bucket * HEADS + h];
}

// ---------------------------------------------------------------------------
// GEMM: C[4096,1024] = A[4096,1024](bf16) @ Wt^T + bias, WMMA bf16
// mode 0: fp32 out row-major (final projection)
// mode 1: Q -> [B,H,S,D] bf16, scaled by 1/8
// mode 2: K -> [B,H,S,D] bf16
// mode 3: V -> [B,H,D,S] bf16 (transposed for attn*V B-fragments)
// 8 waves per block, each wave a 32x32 tile; block tile 128x64.
// ---------------------------------------------------------------------------
__global__ void gemm_bf16_1024(const __bf16* __restrict__ A,
                               const __bf16* __restrict__ Bt,
                               const float* __restrict__ bias,
                               void* __restrict__ out, int mode, float scale) {
  const int K = EMB, N = EMB;
  int lane = threadIdx.x & 31;
  int w = threadIdx.x >> 5;
  int wr = w >> 1, wc = w & 1;
  int m0 = blockIdx.x * 128 + wr * 32;
  int n0 = blockIdx.y * 64 + wc * 32;

  v8f c00 = {}, c01 = {}, c10 = {}, c11 = {};
  const __bf16* a0p = A + (size_t)m0 * K;
  const __bf16* a1p = A + (size_t)(m0 + 16) * K;
  const __bf16* b0p = Bt + (size_t)n0 * K;
  const __bf16* b1p = Bt + (size_t)(n0 + 16) * K;

  for (int kk = 0; kk < K; kk += 32) {
    v16bf a0 = load_a_frag(a0p + kk, K, lane);
    v16bf a1 = load_a_frag(a1p + kk, K, lane);
    v16bf b0 = load_b_frag(b0p + kk, K, lane);
    v16bf b1 = load_b_frag(b1p + kk, K, lane);
    c00 = wmma_bf16(a0, b0, c00);
    c01 = wmma_bf16(a0, b1, c01);
    c10 = wmma_bf16(a1, b0, c10);
    c11 = wmma_bf16(a1, b1, c11);
  }

  int half = lane >> 4, nn = lane & 15;
  auto store_tile = [&](v8f c, int mrow, int ncol) {
    int col = ncol + nn;
    float bc = bias[col];
#pragma unroll
    for (int i = 0; i < 8; ++i) {
      int row = mrow + half * 8 + i;
      float v = c[i] + bc;
      if (mode == 0) {
        ((float*)out)[(size_t)row * N + col] = v;
      } else {
        v *= scale;
        int bb = row >> 11, s = row & (SEQ - 1);
        int hh = col >> 6, d = col & (HDIM - 1);
        if (mode == 3)
          ((__bf16*)out)[(((size_t)bb * HEADS + hh) * HDIM + d) * SEQ + s] = (__bf16)v;
        else
          ((__bf16*)out)[(((size_t)bb * HEADS + hh) * SEQ + s) * HDIM + d] = (__bf16)v;
      }
    }
  };
  store_tile(c00, m0, n0);
  store_tile(c01, m0, n0 + 16);
  store_tile(c10, m0 + 16, n0);
  store_tile(c11, m0 + 16, n0 + 16);
}

// ---------------------------------------------------------------------------
// Attention: one workgroup per (b, h, 16-query tile).
// LDS: fp32 scores [16][2056] + bf16 probs [16][2056] + split-K partials.
// ---------------------------------------------------------------------------
#define SLD 2056   // padded leading dim (banks shift per row)

__global__ void attn_kernel(const __bf16* __restrict__ Qb,
                            const __bf16* __restrict__ Kb,
                            const __bf16* __restrict__ Vt,
                            __bf16* __restrict__ ctxb,
                            const float* __restrict__ tab) {
  extern __shared__ char smem[];
  float*  sc    = (float*)smem;                              // 16*SLD f32
  __bf16* attnb = (__bf16*)(smem + 16 * SLD * 4);            // 16*SLD bf16
  float*  part  = (float*)(smem + 16 * SLD * 4 + 16 * SLD * 2); // 4*32*8 f32

  int lane = threadIdx.x & 31, w = threadIdx.x >> 5;
  int qt = blockIdx.x, h = blockIdx.y, b = blockIdx.z;

  const __bf16* Qh = Qb + (((size_t)b * HEADS + h) * SEQ + (size_t)qt * 16) * HDIM;
  const __bf16* Kh = Kb + (((size_t)b * HEADS + h) * SEQ) * HDIM;
  const __bf16* Vh = Vt + (((size_t)b * HEADS + h) * HDIM) * SEQ;

  // --- Phase 1: scores = (Q/8) @ K^T into LDS ---
  v16bf qa0 = load_a_frag(Qh + 0, HDIM, lane);
  v16bf qa1 = load_a_frag(Qh + 32, HDIM, lane);
  int half = lane >> 4, nn = lane & 15;

  for (int kt = w; kt < SEQ / 16; kt += 8) {
    int n0 = kt * 16;
    v8f c = {};
    v16bf kb0 = load_b_frag(Kh + (size_t)n0 * HDIM + 0, HDIM, lane);
    v16bf kb1 = load_b_frag(Kh + (size_t)n0 * HDIM + 32, HDIM, lane);
    c = wmma_bf16(qa0, kb0, c);
    c = wmma_bf16(qa1, kb1, c);
#pragma unroll
    for (int i = 0; i < 8; ++i)
      sc[(size_t)(half * 8 + i) * SLD + n0 + nn] = c[i];
  }
  __syncthreads();

  // --- Phase 2: bias add + softmax; wave w handles rows 2w, 2w+1 ---
#pragma unroll
  for (int rr = 0; rr < 2; ++rr) {
    int r = w * 2 + rr;
    int qg = qt * 16 + r;
    const float* trow = tab + (size_t)h * TABW + (SEQ - 1) - qg;
    float* srow = sc + (size_t)r * SLD;

    float mx = -3.4e38f;
    for (int j = lane; j < SEQ; j += 32) {
      float x = srow[j] + trow[j];
      srow[j] = x;
      mx = fmaxf(mx, x);
    }
#pragma unroll
    for (int o = 16; o > 0; o >>= 1) mx = fmaxf(mx, __shfl_xor(mx, o, 32));

    float sum = 0.0f;
    for (int j = lane; j < SEQ; j += 32) {
      float e = __expf(srow[j] - mx);
      srow[j] = e;
      sum += e;
    }
#pragma unroll
    for (int o = 16; o > 0; o >>= 1) sum += __shfl_xor(sum, o, 32);

    float inv = 1.0f / sum;
    __bf16* arow = attnb + (size_t)r * SLD;
    for (int j = lane; j < SEQ; j += 32) arow[j] = (__bf16)(srow[j] * inv);
  }
  __syncthreads();

  // --- Phase 3: ctx = attn @ V ; 8 waves = 4 d-tiles x split-K(2) ---
  int dt = w & 3, kh = w >> 2;
  int n0 = dt * 16;
  v8f c = {};
  for (int kc = kh * 32; kc < kh * 32 + 32; ++kc) {
    int kk0 = kc * 32;
    v16bf a  = load_a_frag(attnb + kk0, SLD, lane);            // probs from LDS
    v16bf vb = load_b_frag(Vh + (size_t)n0 * SEQ + kk0, SEQ, lane);
    c = wmma_bf16(a, vb, c);
  }
  if (kh == 1) {
#pragma unroll
    for (int i = 0; i < 8; ++i) part[((size_t)dt * 32 + lane) * 8 + i] = c[i];
  }
  __syncthreads();
  if (kh == 0) {
#pragma unroll
    for (int i = 0; i < 8; ++i) {
      float v = c[i] + part[((size_t)dt * 32 + lane) * 8 + i];
      int s = qt * 16 + half * 8 + i;
      int col = h * HDIM + n0 + nn;
      ctxb[((size_t)b * SEQ + s) * EMB + col] = (__bf16)v;
    }
  }
}

// ---------------------------------------------------------------------------
// Host launch
// ---------------------------------------------------------------------------
extern "C" void kernel_launch(void* const* d_in, const int* in_sizes, int n_in,
                              void* d_out, int out_size, void* d_ws, size_t ws_size,
                              hipStream_t stream) {
  const float* hs  = (const float*)d_in[0];
  const float* Wq  = (const float*)d_in[1];
  const float* bq  = (const float*)d_in[2];
  const float* Wk  = (const float*)d_in[3];
  const float* bk  = (const float*)d_in[4];
  const float* Wv  = (const float*)d_in[5];
  const float* bv  = (const float*)d_in[6];
  const float* Wo  = (const float*)d_in[7];
  const float* bo  = (const float*)d_in[8];
  const float* rel = (const float*)d_in[9];

  char* ws = (char*)d_ws;
  const size_t XB_OFF  = 0;                                   // 4096*1024 bf16 = 8 MB
  const size_t WT_OFF  = XB_OFF + (size_t)MROWS * EMB * 2;    // 4x 1024*1024 bf16 = 8 MB
  const size_t QB_OFF  = WT_OFF + 4ull * EMB * EMB * 2;       // 8 MB
  const size_t KB_OFF  = QB_OFF + (size_t)MROWS * EMB * 2;    // 8 MB
  const size_t VT_OFF  = KB_OFF + (size_t)MROWS * EMB * 2;    // 8 MB
  const size_t CX_OFF  = VT_OFF + (size_t)MROWS * EMB * 2;    // 8 MB
  const size_t TAB_OFF = CX_OFF + (size_t)MROWS * EMB * 2;    // 16*4095 f32

  __bf16* Xb  = (__bf16*)(ws + XB_OFF);
  __bf16* Wtq = (__bf16*)(ws + WT_OFF + 0ull * EMB * EMB * 2);
  __bf16* Wtk = (__bf16*)(ws + WT_OFF + 1ull * EMB * EMB * 2);
  __bf16* Wtv = (__bf16*)(ws + WT_OFF + 2ull * EMB * EMB * 2);
  __bf16* Wto = (__bf16*)(ws + WT_OFF + 3ull * EMB * EMB * 2);
  __bf16* Qb  = (__bf16*)(ws + QB_OFF);
  __bf16* Kb  = (__bf16*)(ws + KB_OFF);
  __bf16* Vt  = (__bf16*)(ws + VT_OFF);
  __bf16* Ctx = (__bf16*)(ws + CX_OFF);
  float*  tab = (float*)(ws + TAB_OFF);

  // Prep: convert activations, transpose+convert weights, bias table.
  {
    int n = MROWS * EMB;
    convert_f32_to_bf16<<<(n + 255) / 256, 256, 0, stream>>>(hs, Xb, n);
  }
  transpose_w_bf16<<<dim3(32, 32), dim3(32, 8), 0, stream>>>(Wq, Wtq);
  transpose_w_bf16<<<dim3(32, 32), dim3(32, 8), 0, stream>>>(Wk, Wtk);
  transpose_w_bf16<<<dim3(32, 32), dim3(32, 8), 0, stream>>>(Wv, Wtv);
  transpose_w_bf16<<<dim3(32, 32), dim3(32, 8), 0, stream>>>(Wo, Wto);
  {
    int n = HEADS * TABW;
    build_bias_table<<<(n + 255) / 256, 256, 0, stream>>>(rel, tab);
  }

  // QKV projections (Q pre-scaled by 1/sqrt(64)).
  dim3 gg(MROWS / 128, EMB / 64);
  gemm_bf16_1024<<<gg, 256, 0, stream>>>(Xb, Wtq, bq, (void*)Qb, 1, 0.125f);
  gemm_bf16_1024<<<gg, 256, 0, stream>>>(Xb, Wtk, bk, (void*)Kb, 2, 1.0f);
  gemm_bf16_1024<<<gg, 256, 0, stream>>>(Xb, Wtv, bv, (void*)Vt, 3, 1.0f);

  // Attention with LDS-resident score rows (16 x 2048 per workgroup).
  size_t smem = (size_t)16 * SLD * 4 + (size_t)16 * SLD * 2 + 4 * 32 * 8 * 4;
  attn_kernel<<<dim3(SEQ / 16, HEADS, BATCH), 256, smem, stream>>>(Qb, Kb, Vt, Ctx, tab);

  // Output projection -> fp32 d_out.
  gemm_bf16_1024<<<gg, 256, 0, stream>>>(Ctx, Wto, bo, d_out, 0, 1.0f);
}
